// DynamicFlowAttention_90417651515905
// MI455X (gfx1250) — matrix operand order repaired and checked by reading, not compiled, not run on hardware
//
#include <hip/hip_runtime.h>
#include <hip/hip_bf16.h>

// Problem constants (match reference)
#define B_    2
#define N_    4096
#define DIM_  512
#define PDIM_ 64
#define K_    16
#define CH_ROWS 128                   // m-rows staged per TDM chunk (32 KB)

typedef __attribute__((ext_vector_type(16))) __bf16 v16bf;
typedef __attribute__((ext_vector_type(8)))  float  v8f;
typedef __attribute__((ext_vector_type(4)))  unsigned u32x4;
typedef __attribute__((ext_vector_type(8)))  unsigned u32x8;

union BF16x16 { v16bf v; uint32_t u[8]; };

// pack two f32 -> packed bf16 pair (low = a, high = b), RNE
static __device__ __forceinline__ uint32_t pkbf(float a, float b) {
#if __has_builtin(__builtin_amdgcn_cvt_pk_bf16_f32)
  typedef __attribute__((ext_vector_type(2))) __bf16 v2bf;
  v2bf t = __builtin_amdgcn_cvt_pk_bf16_f32(a, b);
  return __builtin_bit_cast(uint32_t, t);
#else
  unsigned ua = __builtin_bit_cast(unsigned, a);
  unsigned ub = __builtin_bit_cast(unsigned, b);
  ua += 0x7FFFu + ((ua >> 16) & 1u);
  ub += 0x7FFFu + ((ub >> 16) & 1u);
  return (ua >> 16) | (ub & 0xFFFF0000u);
#endif
}

// async global -> LDS copy, 16 bytes per lane (GLOBAL_LOAD_ASYNC_TO_LDS_B128)
static __device__ __forceinline__ void async_copy_b128(void* lds_dst, const void* gsrc) {
  const unsigned ldsa = (unsigned)(size_t)lds_dst;   // LDS byte address (aperture low bits)
  asm volatile("global_load_async_to_lds_b128 %0, %1, off"
               :: "v"(ldsa), "v"(gsrc)
               : "memory");
}
static __device__ __forceinline__ void wait_async0() {
#if __has_builtin(__builtin_amdgcn_s_wait_asynccnt)
  __builtin_amdgcn_s_wait_asynccnt(0);
#else
  asm volatile("s_wait_asynccnt 0x0" ::: "memory");
#endif
}
static __device__ __forceinline__ void wait_tensor0() {
#if __has_builtin(__builtin_amdgcn_s_wait_tensorcnt)
  __builtin_amdgcn_s_wait_tensorcnt(0);
#else
  asm volatile("s_wait_tensorcnt 0x0" ::: "memory");
#endif
}

// Tensor Data Mover: 2D tile (tile_d0 elems x tile_d1 rows, 4-byte elems) from a
// row-major tensor (row stride stride_elems) at gaddr -> LDS at lds_addr.
// D# per CDNA5 ISA ch.8: group0 = {count|flags, lds_addr, global_addr, type=2},
// group1 = {data_size, tensor_dim0/1, tile_dim0/1/2, tensor_dim0_stride}.
static __device__ __forceinline__ void tdm_load_2d(unsigned lds_addr, const void* gaddr,
                                                   unsigned tile_d0, unsigned tile_d1,
                                                   unsigned tensor_d0, unsigned tensor_d1,
                                                   unsigned stride_elems) {
  const unsigned long long ga = (unsigned long long)(size_t)gaddr;
  u32x4 g0;
  g0[0] = 1u;                                   // count=1, user descriptor
  g0[1] = lds_addr;                             // bits 63:32
  g0[2] = (unsigned)ga;                         // bits 95:64
  g0[3] = (unsigned)(ga >> 32) | (2u << 30);    // addr hi | type=2 ("image")
  u32x8 g1;
  g1[0] = 2u << 16;                             // data_size=2 (4 bytes); no multicast
  g1[1] = (tensor_d0 & 0xFFFFu) << 16;          // tensor_dim0 [79:48] low part
  g1[2] = (tensor_d0 >> 16) | ((tensor_d1 & 0xFFFFu) << 16);   // dim0 hi | dim1 lo
  g1[3] = (tensor_d1 >> 16) | (tile_d0 << 16);  // dim1 hi | tile_dim0 [127:112]
  g1[4] = tile_d1;                              // tile_dim1 [143:128]; tile_dim2=0
  g1[5] = stride_elems;                         // tensor_dim0_stride [207:160] low
  g1[6] = 0u;
  g1[7] = 0u;
  asm volatile("tensor_load_to_lds %0, %1" :: "s"(g0), "s"(g1) : "memory");
}

// Build A fragment (16x32 bf16): lane<16 K=[0..7],[16..23]; lane>=16 K=[8..15],[24..31]
static __device__ __forceinline__ v16bf pack_a(const float* ap) {
  const float4* p4 = (const float4*)ap;
  float4 x0 = p4[0], x1 = p4[1], x2 = p4[4], x3 = p4[5];
  BF16x16 t;
  t.u[0] = pkbf(x0.x, x0.y); t.u[1] = pkbf(x0.z, x0.w);
  t.u[2] = pkbf(x1.x, x1.y); t.u[3] = pkbf(x1.z, x1.w);
  t.u[4] = pkbf(x2.x, x2.y); t.u[5] = pkbf(x2.z, x2.w);
  t.u[6] = pkbf(x3.x, x3.y); t.u[7] = pkbf(x3.z, x3.w);
  return t.v;
}
// Build B fragment (32x16 bf16): col = lane&15; lane<16 K=[0..7],[16..23]; lane>=16 K=[8..15],[24..31]
static __device__ __forceinline__ v16bf pack_b(const float* bp) {
  const float4* p4 = (const float4*)bp;
  float4 y0 = p4[0], y1 = p4[1], y2 = p4[4], y3 = p4[5];
  BF16x16 t;
  t.u[0] = pkbf(y0.x, y0.y); t.u[1] = pkbf(y0.z, y0.w);
  t.u[2] = pkbf(y1.x, y1.y); t.u[3] = pkbf(y1.z, y1.w);
  t.u[4] = pkbf(y2.x, y2.y); t.u[5] = pkbf(y2.z, y2.w);
  t.u[6] = pkbf(y3.x, y3.y); t.u[7] = pkbf(y3.z, y3.w);
  return t.v;
}

// ---------------------------------------------------------------------------
// Kernel 1: flow projection + position update. One wave per 16x16 tile.
// ---------------------------------------------------------------------------
__global__ void flow_kernel(const float* __restrict__ states,
                            const float* __restrict__ Wf,
                            const float* __restrict__ bfl,
                            const float* __restrict__ positions,
                            float* __restrict__ flow_out,
                            float* __restrict__ newpos_out) {
  const int lane = threadIdx.x & 31;
  const int l16  = lane & 15;
  const int hf   = lane >> 4;
  const int row0 = blockIdx.x * 16;    // over B*N
  const int col0 = blockIdx.y * 16;    // over PDIM

  v8f acc = {};
  const float* arow = states + (size_t)(row0 + l16) * DIM_;
  const float* brow = Wf     + (size_t)(col0 + l16) * DIM_;
  for (int k0 = 0; k0 < DIM_; k0 += 32) {
    v16bf a  = pack_a(arow + k0 + 8 * hf);
    v16bf bb = pack_b(brow + k0 + 8 * hf);
    acc = __builtin_amdgcn_wmma_f32_16x16x32_bf16(false, a, false, bb,
                                                  (short)0, acc, false, false);
  }
  const int col = col0 + l16;
  const float bias = bfl[col];
#pragma unroll
  for (int r = 0; r < 8; ++r) {        // D: M = r + 8*hf, N = l16
    const int row = row0 + r + 8 * hf;
    const float f = acc[r] + bias;
    flow_out  [(size_t)row * PDIM_ + col] = f;
    newpos_out[(size_t)row * PDIM_ + col] =
        positions[(size_t)row * PDIM_ + col] + 0.1f * f;
  }
}

// ---------------------------------------------------------------------------
// Kernel 2: value projection.
// ---------------------------------------------------------------------------
__global__ void val_kernel(const float* __restrict__ states,
                           const float* __restrict__ Wv,
                           const float* __restrict__ bv,
                           float* __restrict__ values) {
  const int lane = threadIdx.x & 31;
  const int l16  = lane & 15;
  const int hf   = lane >> 4;
  const int row0 = blockIdx.x * 16;
  const int col0 = blockIdx.y * 16;

  v8f acc = {};
  const float* arow = states + (size_t)(row0 + l16) * DIM_;
  const float* brow = Wv     + (size_t)(col0 + l16) * DIM_;
  for (int k0 = 0; k0 < DIM_; k0 += 32) {
    v16bf a  = pack_a(arow + k0 + 8 * hf);
    v16bf bb = pack_b(brow + k0 + 8 * hf);
    acc = __builtin_amdgcn_wmma_f32_16x16x32_bf16(false, a, false, bb,
                                                  (short)0, acc, false, false);
  }
  const int col = col0 + l16;
  const float bias = bv[col];
#pragma unroll
  for (int r = 0; r < 8; ++r) {
    const int row = row0 + r + 8 * hf;
    values[(size_t)row * DIM_ + col] = acc[r] + bias;
  }
}

// ---------------------------------------------------------------------------
// Kernel 3: squared norms of new positions
// ---------------------------------------------------------------------------
__global__ void sqn_kernel(const float* __restrict__ np, float* __restrict__ sqn) {
  const int row = blockIdx.x * blockDim.x + threadIdx.x;
  if (row >= B_ * N_) return;
  const float4* p = (const float4*)(np + (size_t)row * PDIM_);
  float s = 0.f;
#pragma unroll
  for (int i = 0; i < PDIM_ / 4; ++i) {
    float4 v = p[i];
    s += v.x * v.x + v.y * v.y + v.z * v.z + v.w * v.w;
  }
  sqn[row] = s;
}

// ---------------------------------------------------------------------------
// Kernel 4: fused distance strip (WMMA Gram fed from TDM/async staged LDS
// panels) + kNN select + weighted aggregate. One 256-thread WG per strip.
// LDS: dist 256KB + B-chunk 32KB (aliased w/ candidate lists) + A-panel 4KB.
// ---------------------------------------------------------------------------
__global__ void attn_kernel(const float* __restrict__ newpos,
                            const float* __restrict__ sqn,
                            const float* __restrict__ values,
                            float* __restrict__ ctx) {
  extern __shared__ float smem[];
  float* dist   = smem;                        // 16 * N_          (65536 f)
  float* bbuf   = dist + 16 * N_;              // CH_ROWS * PDIM_  ( 8192 f)
  float* apanel = bbuf + CH_ROWS * PDIM_;      // 16 * PDIM_       ( 1024 f)
  float* topW   = apanel + 16 * PDIM_;         // 16*K_
  int*   topI   = (int*)(topW + 16 * K_);      // 16*K_
  float* topD   = (float*)(topI + 16 * K_);    // 16*K_
  // candidate lists alias the B staging buffer (only live after phase 1)
  float* candD  = bbuf;                        // 16*16*K_ (4096 f)
  int*   candI  = (int*)(bbuf + 16 * 16 * K_); // 16*16*K_ (4096 i)

  const int b    = blockIdx.x / (N_ / 16);
  const int n0   = (blockIdx.x % (N_ / 16)) * 16;
  const int tid  = threadIdx.x;
  const int wave = tid >> 5;
  const int lane = tid & 31;
  const int l16  = lane & 15;
  const int hf   = lane >> 4;
  const size_t base = (size_t)b * N_;

  // ---- stage A panel (16 rows x 64 f32 = 4 KB) via per-lane async copies ----
  async_copy_b128(apanel + tid * 4,
                  newpos + (base + n0 + (tid >> 4)) * PDIM_ + (tid & 15) * 4);
  wait_async0();
  __syncthreads();

  v16bf afrag[2];
#pragma unroll
  for (int kk = 0; kk < 2; ++kk)
    afrag[kk] = pack_a(apanel + l16 * PDIM_ + kk * 32 + 8 * hf);

  float sq_row[8];
#pragma unroll
  for (int r = 0; r < 8; ++r) sq_row[r] = sqn[base + n0 + r + 8 * hf];

  // ---- phase 1: 32 chunks of 128 m-rows; TDM-stage chunk, 1 tile/wave ----
  const unsigned bbuf_lds = (unsigned)(size_t)bbuf;
  for (int chunk = 0; chunk < N_ / CH_ROWS; ++chunk) {
    const int m0c = chunk * CH_ROWS;
    __syncthreads();                            // previous chunk compute done
    if (wave == 0) {                            // one Tensor DMA per chunk
      tdm_load_2d(bbuf_lds, newpos + (base + m0c) * PDIM_,
                  /*tile_d0=*/PDIM_, /*tile_d1=*/CH_ROWS,
                  /*tensor_d0=*/PDIM_, /*tensor_d1=*/N_,
                  /*stride_elems=*/PDIM_);
      wait_tensor0();
    }
    __syncthreads();                            // chunk visible to all waves

    const int m0 = m0c + wave * 16;
    const int lr = wave * 16 + l16;             // local B row in bbuf
    v8f acc = {};
#pragma unroll
    for (int kk = 0; kk < 2; ++kk) {
      v16bf bb = pack_b(bbuf + lr * PDIM_ + kk * 32 + 8 * hf);
      acc = __builtin_amdgcn_wmma_f32_16x16x32_bf16(false, afrag[kk], false, bb,
                                                    (short)0, acc, false, false);
    }
    const int   mcol = m0 + l16;
    const float sqm  = sqn[base + mcol];
#pragma unroll
    for (int r = 0; r < 8; ++r) {
      const int nrow = n0 + r + 8 * hf;
      float d2 = fmaxf(sq_row[r] + sqm - 2.0f * acc[r], 1e-12f);
      float d  = __builtin_sqrtf(d2);
      if (mcol == nrow) d += 1.0e6f;            // exclude self
      dist[(r + 8 * hf) * N_ + mcol] = d;
    }
  }
  __syncthreads();

  // ---- phase 2a: 16 threads/row, each keeps top-16 of a 256-wide segment ----
  {
    const int row = tid >> 4, seg = tid & 15;
    float* md = candD + tid * K_;
    int*   mi = candI + tid * K_;
#pragma unroll
    for (int s = 0; s < K_; ++s) { md[s] = 3.0e38f; mi[s] = -1; }
    const float* drow = dist + row * N_ + seg * (N_ / 16);
    for (int j = 0; j < N_ / 16; ++j) {
      const float dv = drow[j];
      if (dv < md[K_ - 1]) {
        int p = K_ - 1;
        while (p > 0 && md[p - 1] > dv) { md[p] = md[p - 1]; mi[p] = mi[p - 1]; --p; }
        md[p] = dv; mi[p] = seg * (N_ / 16) + j;
      }
    }
  }
  __syncthreads();

  // ---- phase 2b: merge 16 candidate lists per row; Gaussian weights ----
  if (tid < 16) {
    float* td = topD + tid * K_;
    int*   ti = topI + tid * K_;
#pragma unroll
    for (int s = 0; s < K_; ++s) { td[s] = 3.0e38f; ti[s] = -1; }
    const float* cd = candD + tid * 16 * K_;
    const int*   ci = candI + tid * 16 * K_;
    for (int j = 0; j < 16 * K_; ++j) {
      const float dv = cd[j];
      if (dv < td[K_ - 1]) {
        int p = K_ - 1;
        while (p > 0 && td[p - 1] > dv) { td[p] = td[p - 1]; ti[p] = ti[p - 1]; --p; }
        td[p] = dv; ti[p] = ci[j];
      }
    }
    float* tw = topW + tid * K_;
    float wsum = 0.f;
#pragma unroll
    for (int j = 0; j < K_; ++j) {               // exp(-d/(2*sigma^2)), sigma=1
      const float w = __expf(-0.5f * td[j]);
      tw[j] = w; wsum += w;
    }
    const float inv = 1.0f / (wsum + 1e-8f);
#pragma unroll
    for (int j = 0; j < K_; ++j) tw[j] *= inv;
  }
  __syncthreads();

  // ---- phase 3: context[row,:] = sum_j w_j * values[nbr_j,:] ----
  {
    const int row = tid >> 4;
    const int off = (tid & 15) * (DIM_ / 16);    // 32 dims per thread
    float acc[DIM_ / 16];
#pragma unroll
    for (int c = 0; c < DIM_ / 16; ++c) acc[c] = 0.f;
    const float* tw = topW + row * K_;
    const int*   ti = topI + row * K_;
    for (int j = 0; j < K_; ++j) {
      const float w = tw[j];
      const int   m = ti[j];
      const float4* vp = (const float4*)(values + (base + m) * DIM_ + off);
#pragma unroll
      for (int c4 = 0; c4 < DIM_ / 64; ++c4) {
        float4 v = vp[c4];
        acc[c4 * 4 + 0] += w * v.x;
        acc[c4 * 4 + 1] += w * v.y;
        acc[c4 * 4 + 2] += w * v.z;
        acc[c4 * 4 + 3] += w * v.w;
      }
    }
    float* cp = ctx + (base + n0 + row) * DIM_ + off;
#pragma unroll
    for (int c = 0; c < DIM_ / 16; ++c) cp[c] = acc[c];
  }
}

// ---------------------------------------------------------------------------
extern "C" void kernel_launch(void* const* d_in, const int* in_sizes, int n_in,
                              void* d_out, int out_size, void* d_ws, size_t ws_size,
                              hipStream_t stream) {
  (void)in_sizes; (void)n_in; (void)out_size; (void)ws_size;
  const float* states    = (const float*)d_in[0];
  const float* positions = (const float*)d_in[1];
  const float* Wf        = (const float*)d_in[2];
  const float* bfl       = (const float*)d_in[3];
  const float* Wv        = (const float*)d_in[4];
  const float* bv        = (const float*)d_in[5];

  float* ctx    = (float*)d_out;                       // [B,N,DIM]
  float* newpos = ctx    + (size_t)B_ * N_ * DIM_;     // [B,N,PDIM]
  float* flow   = newpos + (size_t)B_ * N_ * PDIM_;    // [B,N,PDIM]

  float* values = (float*)d_ws;                        // [B,N,DIM]  (16 MB)
  float* sqn    = values + (size_t)B_ * N_ * DIM_;     // [B,N]

  flow_kernel<<<dim3((B_ * N_) / 16, PDIM_ / 16), dim3(32), 0, stream>>>(
      states, Wf, bfl, positions, flow, newpos);
  val_kernel<<<dim3((B_ * N_) / 16, DIM_ / 16), dim3(32), 0, stream>>>(
      states, Wv, bv, values);
  sqn_kernel<<<dim3((B_ * N_) / 256), dim3(256), 0, stream>>>(newpos, sqn);

  const size_t smem_bytes =
      (size_t)(16 * N_              /* dist strip      */
             + CH_ROWS * PDIM_      /* B chunk / cand  */
             + 16 * PDIM_           /* A panel         */
             + 16 * K_ * 3          /* topW/topI/topD  */) * sizeof(float);
  attn_kernel<<<dim3(B_ * (N_ / 16)), dim3(256), smem_bytes, stream>>>(
      newpos, sqn, values, ctx);
}
